// AttentionBlock_35811437314650
// MI455X (gfx1250) — matrix operand order repaired and checked by reading, not compiled, not run on hardware
//
#include <hip/hip_runtime.h>
#include <hip/hip_bf16.h>

// ---------------------------------------------------------------------------
// MI455X (gfx1250) transformer block: bf16 WMMA everywhere, f32 accumulation.
// Weights pre-converted to bf16 once (fits L2); all GEMMs double-buffered.
// ---------------------------------------------------------------------------

typedef __bf16 v16bf __attribute__((ext_vector_type(16)));
typedef __bf16 v8bf  __attribute__((ext_vector_type(8)));
typedef __bf16 v4bf  __attribute__((ext_vector_type(4)));
typedef __bf16 v2bf  __attribute__((ext_vector_type(2)));
typedef float  v8f   __attribute__((ext_vector_type(8)));

#define D_MODEL 2048
#define NHEAD   16
#define DHEAD   128
#define DROT    64
#define FFN_DIM 8192
#define SEQ     2048
#define BATCH   2
#define ROWS    (BATCH * SEQ)   // 4096

static constexpr float kEps   = 1e-6f;
static constexpr float kScale = 0.07216878364870323f;  // 1/sqrt(128+64)
static constexpr float kTheta = 1000000.0f;
static constexpr float kNegBig = -1e30f;

// Let the allocator use the whole VGPR file: WMMA accumulators must stay
// resident (no scratch spills in the inner loop).
#define WMMA_ATTR_256 __attribute__((amdgpu_flat_work_group_size(256, 256), amdgpu_waves_per_eu(1)))
#define WMMA_ATTR_128 __attribute__((amdgpu_flat_work_group_size(128, 128), amdgpu_waves_per_eu(1)))

// ---------------------------------------------------------------------------
// Fragment loaders per CDNA5 ISA 7.12.2 (wave32).
// A 16x32 bf16: lanes 0-15 = rows M=0..15 holding K 0-7 (v0-3) & 16-23 (v4-7);
//               lanes 16-31 = same rows holding K 8-15 & 24-31.
// B 32x16 bf16: lanes 0-15 = cols N=0..15 holding K 0-7 (v0-3) & 16-23 (v4-7);
//               lanes 16-31 = same cols holding K 8-15 & 24-31.
// ---------------------------------------------------------------------------
__device__ __forceinline__ v16bf load_frag_a(const __bf16* base, int stride, int kbase) {
    const int lane = threadIdx.x & 31;
    const int hl   = lane >> 4;
    const int r    = lane & 15;
    const __bf16* p = base + r * stride + kbase + hl * 8;
    v8bf lo = *(const v8bf*)(p);        // K = kbase + hl*8 + 0..7      -> v0-3
    v8bf hi = *(const v8bf*)(p + 16);   // K = kbase + 16 + hl*8 + 0..7 -> v4-7
    v16bf o;
#pragma unroll
    for (int i = 0; i < 8; ++i) { o[i] = lo[i]; o[i + 8] = hi[i]; }
    return o;
}

__device__ __forceinline__ v16bf load_frag_b(const __bf16* base, int stride, int kbase) {
    const int lane = threadIdx.x & 31;
    const int hl   = lane >> 4;
    const int n    = lane & 15;
    const __bf16* p = base + n * stride + kbase + hl * 8;
    v8bf lo = *(const v8bf*)(p);        // K = kbase + hl*8 + 0..7      -> v0-3
    v8bf hi = *(const v8bf*)(p + 16);   // K = kbase + 16 + hl*8 + 0..7 -> v4-7
    v16bf o;
#pragma unroll
    for (int i = 0; i < 8; ++i) { o[i] = lo[i]; o[i + 8] = hi[i]; }
    return o;
}

__device__ __forceinline__ v8f wmma_bf16(v16bf a, v16bf b, v8f c) {
    return __builtin_amdgcn_wmma_f32_16x16x32_bf16(false, a, false, b,
                                                   (short)0, c, false, false);
}

__device__ __forceinline__ v8f v8f_zero() {
    v8f z;
#pragma unroll
    for (int i = 0; i < 8; ++i) z[i] = 0.0f;
    return z;
}

// ---------------------------------------------------------------------------
// fp32 -> bf16 bulk convert (weights; element counts are multiples of 1024).
// ---------------------------------------------------------------------------
__global__ void mla_cvt_bf16_kernel(const float* __restrict__ in,
                                    __bf16* __restrict__ out) {
    const size_t i = ((size_t)blockIdx.x * blockDim.x + threadIdx.x) * 4;
    v2bf p0; p0[0] = (__bf16)in[i];     p0[1] = (__bf16)in[i + 1];
    v2bf p1; p1[0] = (__bf16)in[i + 2]; p1[1] = (__bf16)in[i + 3];
    *(v2bf*)(out + i)     = p0;
    *(v2bf*)(out + i + 2) = p1;
}

// ---------------------------------------------------------------------------
// RMSNorm: one row (D_MODEL=2048) per 256-thread block; fp32 in, bf16 out.
// ---------------------------------------------------------------------------
__global__ void mla_rmsnorm_kernel(const float* __restrict__ x,
                                   const float* __restrict__ w,
                                   __bf16* __restrict__ out) {
    const int row = blockIdx.x;
    const int t   = threadIdx.x;
    const float* xr = x + (size_t)row * D_MODEL;
    float v[8];
    float ss = 0.0f;
#pragma unroll
    for (int i = 0; i < 8; ++i) { v[i] = xr[t + i * 256]; ss += v[i] * v[i]; }
#pragma unroll
    for (int off = 16; off >= 1; off >>= 1) ss += __shfl_xor(ss, off, 32);
    __shared__ float wsum[8];
    if ((t & 31) == 0) wsum[t >> 5] = ss;
    __syncthreads();
    float tot = 0.0f;
#pragma unroll
    for (int i = 0; i < 8; ++i) tot += wsum[i];
    const float inv = rsqrtf(tot / (float)D_MODEL + kEps);
    __bf16* orow = out + (size_t)row * D_MODEL;
#pragma unroll
    for (int i = 0; i < 8; ++i)
        orow[t + i * 256] = (__bf16)(v[i] * inv * w[t + i * 256]);
}

// ---------------------------------------------------------------------------
// RoPE (interleaved pairs), in-place on bf16 [ROWS][NHEAD*DROT].
// ---------------------------------------------------------------------------
__global__ void mla_rope_kernel(__bf16* __restrict__ R) {
    const size_t p = (size_t)blockIdx.x * blockDim.x + threadIdx.x;
    const int ppr = NHEAD * (DROT / 2);           // pairs per row
    const int row = (int)(p / ppr);
    const int c   = (int)(p % ppr);
    const int h   = c >> 5;                        // head
    const int i   = c & 31;                        // pair index within head
    const int pos = row & (SEQ - 1);
    const float freq = __expf(-((float)(2 * i) / (float)DROT) * __logf(kTheta));
    float s, cz;
    __sincosf((float)pos * freq, &s, &cz);
    __bf16* q = R + (size_t)row * (NHEAD * DROT) + h * DROT + 2 * i;
    const float x1 = (float)q[0];
    const float x2 = (float)q[1];
    q[0] = (__bf16)(x1 * cz - x2 * s);
    q[1] = (__bf16)(x2 * cz + x1 * s);
}

// ---------------------------------------------------------------------------
// 128x128-tile GEMM, double-buffered LDS, single barrier per K-step.
// A bf16 [M,K] row-major; B bf16 [K,N] row-major (transposed into N-major LDS
// as packed k-pairs -> ds_store_b32). 256 threads = 8 waves; wave = 16-row x
// 128-col strip = 8 WMMA per 32-wide K step.
// EPI=0: bf16 out. EPI=1: f32 out = acc + Res (residual).
// ---------------------------------------------------------------------------
template <int EPI>
__global__ WMMA_ATTR_256
void mla_gemm128_kernel(const __bf16* __restrict__ A,
                        const __bf16* __restrict__ B,
                        const float* __restrict__ Res,
                        void* __restrict__ Out,
                        int M, int N, int K) {
    __shared__ __align__(16) __bf16 As[2][128 * 32];   // [row][k]
    __shared__ __align__(16) __bf16 Bs[2][128 * 32];   // [n][k] (transposed)
    const int nb   = blockIdx.x * 128;
    const int mb   = blockIdx.y * 128;
    const int t    = threadIdx.x;
    const int wid  = t >> 5;
    const int lane = t & 31;
    const int hl   = lane >> 4;
    const int l15  = lane & 15;
    const int wrow = wid * 16;
    // staging assignments
    const int arow = t >> 1, ac0 = (t & 1) * 16;   // A: 16 bf16 per thread
    const int kp   = t >> 4, n0  = (t & 15) * 8;   // B: 2 rows x 8 cols

    v8bf ra0, ra1, rb0, rb1;
    auto gload = [&](int kb) {
        const __bf16* asrc = A + (size_t)(mb + arow) * K + kb + ac0;
        ra0 = *(const v8bf*)(asrc);
        ra1 = *(const v8bf*)(asrc + 8);
        const __bf16* bsrc = B + (size_t)(kb + 2 * kp) * N + nb + n0;
        rb0 = *(const v8bf*)(bsrc);
        rb1 = *(const v8bf*)(bsrc + N);
    };
    auto lstore = [&](int buf) {
        *(v8bf*)&As[buf][arow * 32 + ac0]     = ra0;
        *(v8bf*)&As[buf][arow * 32 + ac0 + 8] = ra1;
#pragma unroll
        for (int i = 0; i < 8; ++i) {
            v2bf pk; pk[0] = rb0[i]; pk[1] = rb1[i];
            *(v2bf*)&Bs[buf][(n0 + i) * 32 + 2 * kp] = pk;
        }
    };

    v8f acc[8];
#pragma unroll
    for (int fr = 0; fr < 8; ++fr) acc[fr] = v8f_zero();

    gload(0);
    lstore(0);
    __syncthreads();

    int buf = 0;
    for (int kb = 0; kb < K; kb += 32) {
        if (kb + 32 < K) {               // stage next tile into the other buffer
            gload(kb + 32);
            lstore(buf ^ 1);
            __builtin_prefetch(B + (size_t)(kb + 64 + 2 * kp) * N + nb, 0, 1);
        }
        v16bf a = load_frag_a(&As[buf][wrow * 32], 32, 0);
#pragma unroll
        for (int fr = 0; fr < 8; ++fr) {
            v16bf bf = load_frag_b(&Bs[buf][fr * 16 * 32], 32, 0);
            acc[fr] = wmma_bf16(a, bf, acc[fr]);
        }
        __syncthreads();                 // publishes buf^1, retires reads of buf
        buf ^= 1;
    }

#pragma unroll
    for (int fr = 0; fr < 8; ++fr) {
#pragma unroll
        for (int j = 0; j < 8; ++j) {
            const int row = mb + wrow + j + 8 * hl;
            const int col = nb + fr * 16 + l15;
            const size_t idx = (size_t)row * N + col;
            if (EPI == 0) ((__bf16*)Out)[idx] = (__bf16)acc[fr][j];
            else          ((float*)Out)[idx]  = acc[fr][j] + Res[idx];
        }
    }
}

// ---------------------------------------------------------------------------
// Fused SwiGLU gate/up GEMM, 128x64 tile, double-buffered.
// h = silu(g)*u -> bf16.
// ---------------------------------------------------------------------------
__global__ WMMA_ATTR_256
void mla_ffn_gateup_kernel(const __bf16* __restrict__ A,
                           const __bf16* __restrict__ Bg,
                           const __bf16* __restrict__ Bu,
                           __bf16* __restrict__ H,
                           int M, int N, int K) {
    __shared__ __align__(16) __bf16 As[2][128 * 32];
    __shared__ __align__(16) __bf16 Bsg[2][64 * 32];
    __shared__ __align__(16) __bf16 Bsu[2][64 * 32];
    const int nb   = blockIdx.x * 64;
    const int mb   = blockIdx.y * 128;
    const int t    = threadIdx.x;
    const int wid  = t >> 5;
    const int lane = t & 31;
    const int hl   = lane >> 4;
    const int l15  = lane & 15;
    const int wrow = wid * 16;
    const int arow = t >> 1, ac0 = (t & 1) * 16;
    const int kp   = t & 15, n0  = (t >> 4) * 4;   // B: 2 rows x 4 cols x2 mats

    v8bf ra0, ra1;
    v4bf rg0, rg1, ru0, ru1;
    auto gload = [&](int kb) {
        const __bf16* asrc = A + (size_t)(mb + arow) * K + kb + ac0;
        ra0 = *(const v8bf*)(asrc);
        ra1 = *(const v8bf*)(asrc + 8);
        const __bf16* gsrc = Bg + (size_t)(kb + 2 * kp) * N + nb + n0;
        rg0 = *(const v4bf*)(gsrc);
        rg1 = *(const v4bf*)(gsrc + N);
        const __bf16* usrc = Bu + (size_t)(kb + 2 * kp) * N + nb + n0;
        ru0 = *(const v4bf*)(usrc);
        ru1 = *(const v4bf*)(usrc + N);
    };
    auto lstore = [&](int buf) {
        *(v8bf*)&As[buf][arow * 32 + ac0]     = ra0;
        *(v8bf*)&As[buf][arow * 32 + ac0 + 8] = ra1;
#pragma unroll
        for (int i = 0; i < 4; ++i) {
            v2bf pg; pg[0] = rg0[i]; pg[1] = rg1[i];
            *(v2bf*)&Bsg[buf][(n0 + i) * 32 + 2 * kp] = pg;
            v2bf pu; pu[0] = ru0[i]; pu[1] = ru1[i];
            *(v2bf*)&Bsu[buf][(n0 + i) * 32 + 2 * kp] = pu;
        }
    };

    v8f accg[4], accu[4];
#pragma unroll
    for (int fr = 0; fr < 4; ++fr) { accg[fr] = v8f_zero(); accu[fr] = v8f_zero(); }

    gload(0);
    lstore(0);
    __syncthreads();

    int buf = 0;
    for (int kb = 0; kb < K; kb += 32) {
        if (kb + 32 < K) {
            gload(kb + 32);
            lstore(buf ^ 1);
        }
        v16bf a = load_frag_a(&As[buf][wrow * 32], 32, 0);
#pragma unroll
        for (int fr = 0; fr < 4; ++fr) {
            v16bf bg = load_frag_b(&Bsg[buf][fr * 16 * 32], 32, 0);
            accg[fr] = wmma_bf16(a, bg, accg[fr]);
            v16bf bu = load_frag_b(&Bsu[buf][fr * 16 * 32], 32, 0);
            accu[fr] = wmma_bf16(a, bu, accu[fr]);
        }
        __syncthreads();
        buf ^= 1;
    }

#pragma unroll
    for (int fr = 0; fr < 4; ++fr) {
#pragma unroll
        for (int j = 0; j < 8; ++j) {
            const int row = mb + wrow + j + 8 * hl;
            const int col = nb + fr * 16 + l15;
            const float g = accg[fr][j];
            const float u = accu[fr][j];
            const float s = g / (1.0f + __expf(-g));   // silu
            H[(size_t)row * N + col] = (__bf16)(s * u);
        }
    }
}

// ---------------------------------------------------------------------------
// Flash attention: block = (b,h, 64 query rows); 128 threads = 4 waves of
// 16 rows each. N iterated in 64-token chunks. Scores via WMMA over
// D_H (4 k-steps) + D_R (2 k-steps); online softmax; P -> LDS bf16 ->
// P·V via WMMA into 16x128 f32 accumulators.
// ---------------------------------------------------------------------------
__global__ WMMA_ATTR_128
void mla_attn_kernel(const __bf16* __restrict__ QC,
                     const __bf16* __restrict__ QR,
                     const __bf16* __restrict__ KC,
                     const __bf16* __restrict__ KR,
                     const __bf16* __restrict__ V,
                     __bf16* __restrict__ AO) {
    extern __shared__ __align__(16) char smem[];
    __bf16* sQc = (__bf16*)smem;          // [64][128]
    __bf16* sQr = sQc + 64 * 128;         // [64][64]
    __bf16* sKc = sQr + 64 * 64;          // [64][128]  (token-major)
    __bf16* sKr = sKc + 64 * 128;         // [64][64]
    __bf16* sVt = sKr + 64 * 64;          // [128][64]  (d-major, transposed)
    __bf16* sP  = sVt + 128 * 64;         // 4 waves x [16][64]

    const int bh  = blockIdx.x;
    const int b   = bh >> 4;
    const int h   = bh & 15;
    const int mb  = blockIdx.y * 64;
    const int t   = threadIdx.x;          // 0..127
    const int wid = t >> 5;
    const int lane = t & 31;
    const int hl  = lane >> 4;
    const int l15 = lane & 15;
    const int wrow = wid * 16;

    {   // stage Q tiles
        const int row = t >> 1;
        const int c0  = (t & 1) * 64;
        const __bf16* sq = QC + ((size_t)(b * SEQ + mb + row)) * (NHEAD * DHEAD) + h * DHEAD + c0;
#pragma unroll
        for (int i = 0; i < 8; ++i)
            *(v8bf*)&sQc[row * 128 + c0 + 8 * i] = *(const v8bf*)(sq + 8 * i);
        const int c1 = (t & 1) * 32;
        const __bf16* sr = QR + ((size_t)(b * SEQ + mb + row)) * (NHEAD * DROT) + h * DROT + c1;
#pragma unroll
        for (int i = 0; i < 4; ++i)
            *(v8bf*)&sQr[row * 64 + c1 + 8 * i] = *(const v8bf*)(sr + 8 * i);
    }

    float rm[8], rl[8];
    v8f Of[8];
#pragma unroll
    for (int j = 0; j < 8; ++j) { rm[j] = kNegBig; rl[j] = 0.0f; }
#pragma unroll
    for (int fo = 0; fo < 8; ++fo) Of[fo] = v8f_zero();

    for (int n0 = 0; n0 < SEQ; n0 += 64) {
        __syncthreads();   // previous chunk fully consumed (also covers Q staging)
        {   // stage K/V chunk
            const int row = t >> 1;          // token within chunk
            const int c0  = (t & 1) * 64;
            const size_t tokbase = (size_t)(b * SEQ + n0 + row);
            const __bf16* sk = KC + tokbase * (NHEAD * DHEAD) + h * DHEAD + c0;
#pragma unroll
            for (int i = 0; i < 8; ++i)
                *(v8bf*)&sKc[row * 128 + c0 + 8 * i] = *(const v8bf*)(sk + 8 * i);
            const int c1 = (t & 1) * 32;
            const __bf16* skr = KR + tokbase * (NHEAD * DROT) + h * DROT + c1;
#pragma unroll
            for (int i = 0; i < 4; ++i)
                *(v8bf*)&sKr[row * 64 + c1 + 8 * i] = *(const v8bf*)(skr + 8 * i);
            const __bf16* sv = V + tokbase * (NHEAD * DHEAD) + h * DHEAD + c0;
#pragma unroll
            for (int i8 = 0; i8 < 8; ++i8) {
                v8bf vv = *(const v8bf*)(sv + 8 * i8);
#pragma unroll
                for (int i = 0; i < 8; ++i)
                    sVt[(c0 + 8 * i8 + i) * 64 + row] = vv[i];
            }
        }
        __syncthreads();

        // ---- scores S = Qc·Kc^T + Qr·Kr^T (16 rows x 64 tokens per wave)
        v8f S[4];
#pragma unroll
        for (int fr = 0; fr < 4; ++fr) S[fr] = v8f_zero();
#pragma unroll
        for (int kc = 0; kc < 128; kc += 32) {
            v16bf a = load_frag_a(&sQc[wrow * 128], 128, kc);
#pragma unroll
            for (int fr = 0; fr < 4; ++fr)
                S[fr] = wmma_bf16(a, load_frag_b(&sKc[fr * 16 * 128], 128, kc), S[fr]);
        }
#pragma unroll
        for (int kc = 0; kc < 64; kc += 32) {
            v16bf a = load_frag_a(&sQr[wrow * 64], 64, kc);
#pragma unroll
            for (int fr = 0; fr < 4; ++fr)
                S[fr] = wmma_bf16(a, load_frag_b(&sKr[fr * 16 * 64], 64, kc), S[fr]);
        }

        // ---- online softmax. Accum VGPR j <-> row (j + 8*hl); its 16 lanes
        // (within the lane-half) hold the token columns, so half-wide
        // shfl_xor reductions give per-row stats replicated per lane.
        __bf16* Pw = sP + wid * 16 * 64;
#pragma unroll
        for (int j = 0; j < 8; ++j) {
            float mx = kNegBig;
#pragma unroll
            for (int fr = 0; fr < 4; ++fr) mx = fmaxf(mx, S[fr][j] * kScale);
#pragma unroll
            for (int o = 8; o >= 1; o >>= 1) mx = fmaxf(mx, __shfl_xor(mx, o, 32));
            const float mnew = fmaxf(rm[j], mx);
            const float corr = __expf(rm[j] - mnew);
            float rs = 0.0f;
#pragma unroll
            for (int fr = 0; fr < 4; ++fr) {
                const float pv = __expf(S[fr][j] * kScale - mnew);
                S[fr][j] = pv;
                rs += pv;
            }
#pragma unroll
            for (int o = 8; o >= 1; o >>= 1) rs += __shfl_xor(rs, o, 32);
            rl[j] = rl[j] * corr + rs;
            rm[j] = mnew;
#pragma unroll
            for (int fo = 0; fo < 8; ++fo) Of[fo][j] *= corr;
            const int prow = j + 8 * hl;
#pragma unroll
            for (int fr = 0; fr < 4; ++fr)
                Pw[prow * 64 + fr * 16 + l15] = (__bf16)S[fr][j];
        }

        // ---- O += P·V  (same-wave LDS producer/consumer: DS ops in-order)
#pragma unroll
        for (int kc = 0; kc < 64; kc += 32) {
            v16bf a = load_frag_a(Pw, 64, kc);
#pragma unroll
            for (int fo = 0; fo < 8; ++fo)
                Of[fo] = wmma_bf16(a, load_frag_b(&sVt[fo * 16 * 64], 64, kc), Of[fo]);
        }
    }

    float inv[8];
#pragma unroll
    for (int j = 0; j < 8; ++j) inv[j] = 1.0f / rl[j];
#pragma unroll
    for (int fo = 0; fo < 8; ++fo) {
#pragma unroll
        for (int j = 0; j < 8; ++j) {
            const int mrow = mb + wrow + j + 8 * hl;
            AO[((size_t)(b * SEQ + mrow)) * (NHEAD * DHEAD) + h * DHEAD + fo * 16 + l15] =
                (__bf16)(Of[fo][j] * inv[j]);
        }
    }
}

// ---------------------------------------------------------------------------
// Launch
// ---------------------------------------------------------------------------
extern "C" void kernel_launch(void* const* d_in, const int* in_sizes, int n_in,
                              void* d_out, int out_size, void* d_ws, size_t ws_size,
                              hipStream_t stream) {
    (void)in_sizes; (void)n_in; (void)out_size; (void)ws_size;
    const float* query     = (const float*)d_in[0];
    const float* key_value = (const float*)d_in[1];
    const float* g_q       = (const float*)d_in[2];
    const float* g_kv      = (const float*)d_in[3];
    const float* g_ffn     = (const float*)d_in[4];
    const float* w_qc      = (const float*)d_in[5];
    const float* w_kc      = (const float*)d_in[6];
    const float* w_qr      = (const float*)d_in[7];
    const float* w_kr      = (const float*)d_in[8];
    const float* w_v       = (const float*)d_in[9];
    const float* w_o       = (const float*)d_in[10];
    const float* w_gate    = (const float*)d_in[11];
    const float* w_up      = (const float*)d_in[12];
    const float* w_down    = (const float*)d_in[13];
    float* out = (float*)d_out;

    char* ws = (char*)d_ws;
    size_t off = 0;
    auto alloc = [&](size_t bytes) -> char* {
        char* p = ws + off;
        off += (bytes + 255) & ~(size_t)255;
        return p;
    };
    __bf16* qn  = (__bf16*)alloc((size_t)ROWS * D_MODEL * 2);
    __bf16* kvn = (__bf16*)alloc((size_t)ROWS * D_MODEL * 2);
    __bf16* QC  = (__bf16*)alloc((size_t)ROWS * NHEAD * DHEAD * 2);
    __bf16* KC  = (__bf16*)alloc((size_t)ROWS * NHEAD * DHEAD * 2);
    __bf16* Vb  = (__bf16*)alloc((size_t)ROWS * NHEAD * DHEAD * 2);
    __bf16* QR  = (__bf16*)alloc((size_t)ROWS * NHEAD * DROT * 2);
    __bf16* KR  = (__bf16*)alloc((size_t)ROWS * NHEAD * DROT * 2);
    __bf16* AO  = (__bf16*)alloc((size_t)ROWS * NHEAD * DHEAD * 2);
    float*  X   = (float*)alloc((size_t)ROWS * D_MODEL * 4);
    __bf16* XN  = (__bf16*)alloc((size_t)ROWS * D_MODEL * 2);
    __bf16* Hb  = (__bf16*)alloc((size_t)ROWS * FFN_DIM * 2);
    // bf16 weights (converted once per launch; ~143 MB, mostly L2-resident)
    __bf16* wqc = (__bf16*)alloc((size_t)D_MODEL * NHEAD * DHEAD * 2);
    __bf16* wkc = (__bf16*)alloc((size_t)D_MODEL * NHEAD * DHEAD * 2);
    __bf16* wqr = (__bf16*)alloc((size_t)D_MODEL * NHEAD * DROT * 2);
    __bf16* wkr = (__bf16*)alloc((size_t)D_MODEL * NHEAD * DROT * 2);
    __bf16* wv  = (__bf16*)alloc((size_t)D_MODEL * NHEAD * DHEAD * 2);
    __bf16* wo  = (__bf16*)alloc((size_t)NHEAD * DHEAD * D_MODEL * 2);
    __bf16* wg  = (__bf16*)alloc((size_t)D_MODEL * FFN_DIM * 2);
    __bf16* wu  = (__bf16*)alloc((size_t)D_MODEL * FFN_DIM * 2);
    __bf16* wd  = (__bf16*)alloc((size_t)FFN_DIM * D_MODEL * 2);

    auto cvt = [&](const float* src, __bf16* dst, size_t elems) {
        mla_cvt_bf16_kernel<<<(unsigned)(elems / 1024), 256, 0, stream>>>(src, dst);
    };
    cvt(w_qc,   wqc, (size_t)D_MODEL * NHEAD * DHEAD);
    cvt(w_kc,   wkc, (size_t)D_MODEL * NHEAD * DHEAD);
    cvt(w_qr,   wqr, (size_t)D_MODEL * NHEAD * DROT);
    cvt(w_kr,   wkr, (size_t)D_MODEL * NHEAD * DROT);
    cvt(w_v,    wv,  (size_t)D_MODEL * NHEAD * DHEAD);
    cvt(w_o,    wo,  (size_t)NHEAD * DHEAD * D_MODEL);
    cvt(w_gate, wg,  (size_t)D_MODEL * FFN_DIM);
    cvt(w_up,   wu,  (size_t)D_MODEL * FFN_DIM);
    cvt(w_down, wd,  (size_t)FFN_DIM * D_MODEL);

    // 1) RMSNorm inputs -> bf16
    mla_rmsnorm_kernel<<<ROWS, 256, 0, stream>>>(query, g_q, qn);
    mla_rmsnorm_kernel<<<ROWS, 256, 0, stream>>>(key_value, g_kv, kvn);

    // 2) projections
    dim3 g2048(D_MODEL / 128, ROWS / 128);           // (16, 32)
    dim3 g1024((NHEAD * DROT) / 128, ROWS / 128);    // (8, 32)
    mla_gemm128_kernel<0><<<g2048, 256, 0, stream>>>(qn,  wqc, nullptr, QC, ROWS, NHEAD * DHEAD, D_MODEL);
    mla_gemm128_kernel<0><<<g1024, 256, 0, stream>>>(qn,  wqr, nullptr, QR, ROWS, NHEAD * DROT,  D_MODEL);
    mla_gemm128_kernel<0><<<g2048, 256, 0, stream>>>(kvn, wkc, nullptr, KC, ROWS, NHEAD * DHEAD, D_MODEL);
    mla_gemm128_kernel<0><<<g1024, 256, 0, stream>>>(kvn, wkr, nullptr, KR, ROWS, NHEAD * DROT,  D_MODEL);
    mla_gemm128_kernel<0><<<g2048, 256, 0, stream>>>(kvn, wv,  nullptr, Vb, ROWS, NHEAD * DHEAD, D_MODEL);

    // 3) RoPE on Q_R / K_R
    {
        const size_t pairs = (size_t)ROWS * NHEAD * (DROT / 2);
        const int thr = 256;
        const int blk = (int)((pairs + thr - 1) / thr);
        mla_rope_kernel<<<blk, thr, 0, stream>>>(QR);
        mla_rope_kernel<<<blk, thr, 0, stream>>>(KR);
    }

    // 4) flash attention
    {
        dim3 ga(BATCH * NHEAD, SEQ / 64);            // (32, 32)
        const size_t lds = (size_t)(64 * 128 + 64 * 64 + 64 * 128 + 64 * 64 +
                                    128 * 64 + 4 * 16 * 64) * sizeof(__bf16);  // 72 KB
        mla_attn_kernel<<<ga, 128, lds, stream>>>(QC, QR, KC, KR, Vb, AO);
    }

    // 5) output projection + residual -> x (f32)
    mla_gemm128_kernel<1><<<g2048, 256, 0, stream>>>(AO, wo, query, X, ROWS, D_MODEL, NHEAD * DHEAD);

    // 6) FFN
    mla_rmsnorm_kernel<<<ROWS, 256, 0, stream>>>(X, g_ffn, XN);
    dim3 gffn(FFN_DIM / 64, ROWS / 128);             // (128, 32)
    mla_ffn_gateup_kernel<<<gffn, 256, 0, stream>>>(XN, wg, wu, Hb, ROWS, FFN_DIM, D_MODEL);
    mla_gemm128_kernel<1><<<g2048, 256, 0, stream>>>(Hb, wd, X, out, ROWS, D_MODEL, FFN_DIM);
}
